// Hydra_Embedding_Dimensional_Attention_49701361549932
// MI455X (gfx1250) — matrix operand order repaired and checked by reading, not compiled, not run on hardware
//
#include <hip/hip_runtime.h>

typedef __attribute__((ext_vector_type(16))) __bf16 v16bf;
typedef __attribute__((ext_vector_type(8)))  float  v8f;

namespace {
constexpr int Bx  = 64;    // batch
constexpr int Nn  = 196;   // n dim (GEMM K, also proj out cols)
constexpr int Cc  = 768;   // c dim (GEMM M)
constexpr int KT  = 7;     // K tiles of 32 (196 -> 224)
constexpr int MT  = 48;    // M tiles of 16 (768)
constexpr int NT1 = 40;    // GEMM1 out col tiles of 16 (588 -> 640)
constexpr int J1  = 588;   // q|k|v columns
constexpr int NT2 = 16;    // GEMM2 out col tiles of 16 (196 -> 256)

__device__ __forceinline__ v8f wmma_bf16(v16bf a, v16bf b, v8f c) {
  // (neg_a, A, neg_b, B, c_mod, C, reuse_a, reuse_b)
  return __builtin_amdgcn_wmma_f32_16x16x32_bf16(false, a, false, b, (short)0, c,
                                                 false, false);
}
} // namespace

// ---------------------------------------------------------------------------
// Pack x^T (A matrix of GEMM1) into per-lane WMMA fragment order.
// A[m=c][k=n] = x[b][n][c].  ISA 16-bit A 16x32 layout:
//   lane<16 : elems 0..7 -> K {0..7},  elems 8..15 -> K {16..23}
//   lane>=16: elems 0..7 -> K {8..15}, elems 8..15 -> K {24..31}
// ---------------------------------------------------------------------------
__global__ void pack_x_kernel(const float* __restrict__ x, v16bf* __restrict__ Apack) {
  size_t t = (size_t)blockIdx.x * 256 + threadIdx.x;   // Bx*MT*KT*32 threads
  int lane = (int)(t & 31);
  size_t rest = t >> 5;
  int kt = (int)(rest % KT); rest /= KT;
  int mt = (int)(rest % MT);
  int b  = (int)(rest / MT);
  int m     = mt * 16 + (lane & 15);
  int kbase = kt * 32 + ((lane >> 4) * 8);
  v16bf v;
#pragma unroll
  for (int e = 0; e < 16; ++e) {
    int k = kbase + (e < 8 ? e : e + 8);
    float val = (k < Nn) ? x[((size_t)b * Nn + k) * Cc + m] : 0.0f;
    v[e] = (__bf16)val;
  }
  Apack[t] = v;
}

// ---------------------------------------------------------------------------
// Pack [Wq;Wkv] as B matrix (32x16, K x Nout): B[k][j] = Wcomb[j][k].
//   lane<16 : elems 0..7 -> K {0..7}, 8..15 -> K {16..23}; lane>=16: elems 0..7 -> K {8..15}, 8..15 -> K {24..31}
//   column N = lane & 15
// ---------------------------------------------------------------------------
__global__ void pack_w1_kernel(const float* __restrict__ Wq,
                               const float* __restrict__ Wkv,
                               v16bf* __restrict__ Bpack) {
  int t = blockIdx.x * 256 + threadIdx.x;              // NT1*KT*32 = 8960
  if (t >= NT1 * KT * 32) return;
  int lane = t & 31;
  int rest = t >> 5;
  int kt = rest % KT;
  int nt = rest / KT;
  int j     = nt * 16 + (lane & 15);
  int kbase = kt * 32 + ((lane >> 4) * 8);
  v16bf v;
#pragma unroll
  for (int e = 0; e < 16; ++e) {
    int k = kbase + (e < 8 ? e : e + 8);
    float val = 0.0f;
    if (k < Nn && j < J1)
      val = (j < Nn) ? Wq[(size_t)j * Nn + k] : Wkv[(size_t)(j - Nn) * Nn + k];
    v[e] = (__bf16)val;
  }
  Bpack[t] = v;
}

__global__ void pack_wp_kernel(const float* __restrict__ Wproj,
                               v16bf* __restrict__ Bpack) {
  int t = blockIdx.x * 256 + threadIdx.x;              // NT2*KT*32 = 3584
  if (t >= NT2 * KT * 32) return;
  int lane = t & 31;
  int rest = t >> 5;
  int kt = rest % KT;
  int nt = rest / KT;
  int j     = nt * 16 + (lane & 15);
  int kbase = kt * 32 + ((lane >> 4) * 8);
  v16bf v;
#pragma unroll
  for (int e = 0; e < 16; ++e) {
    int k = kbase + (e < 8 ? e : e + 8);
    float val = (k < Nn && j < Nn) ? Wproj[(size_t)j * Nn + k] : 0.0f;
    v[e] = (__bf16)val;
  }
  Bpack[t] = v;
}

// ---------------------------------------------------------------------------
// GEMM1: Y[b] (768 x 588) = Xt[b] @ [Wq;Wkv]^T via bf16 WMMA, f32 accum.
// 8 waves / block, each wave a 32x32 output (2x2 tiles), block = 128M x 64N.
// ---------------------------------------------------------------------------
__global__ void gemm1_kernel(const v16bf* __restrict__ Apack,
                             const v16bf* __restrict__ Bpack,
                             float* __restrict__ Y) {
  const int lane = threadIdx.x & 31;
  const int wave = threadIdx.x >> 5;
  const int mw   = wave & 3;            // 0..3  (M direction)
  const int nw   = wave >> 2;           // 0..1  (N direction)
  const int b    = blockIdx.z;
  const int mt0  = blockIdx.y * 8 + mw * 2;
  const int nt0  = blockIdx.x * 4 + nw * 2;

  v8f acc00 = {}; v8f acc01 = {}; v8f acc10 = {}; v8f acc11 = {};

  const size_t abase0 = ((size_t)(b * MT + mt0)     * KT) * 32 + lane;
  const size_t abase1 = ((size_t)(b * MT + mt0 + 1) * KT) * 32 + lane;
  const size_t bbase0 = ((size_t)(nt0)     * KT) * 32 + lane;
  const size_t bbase1 = ((size_t)(nt0 + 1) * KT) * 32 + lane;

#pragma unroll
  for (int kt = 0; kt < KT; ++kt) {
    v16bf a0 = Apack[abase0 + (size_t)kt * 32];
    v16bf a1 = Apack[abase1 + (size_t)kt * 32];
    v16bf b0 = Bpack[bbase0 + (size_t)kt * 32];
    v16bf b1 = Bpack[bbase1 + (size_t)kt * 32];
    acc00 = wmma_bf16(a0, b0, acc00);
    acc01 = wmma_bf16(a0, b1, acc01);
    acc10 = wmma_bf16(a1, b0, acc10);
    acc11 = wmma_bf16(a1, b1, acc11);
  }

  const int nlo = lane & 15;
  const int mhi = (lane >> 4) * 8;
#pragma unroll
  for (int i = 0; i < 2; ++i) {
#pragma unroll
    for (int j = 0; j < 2; ++j) {
      v8f acc = (i == 0) ? (j == 0 ? acc00 : acc01) : (j == 0 ? acc10 : acc11);
      int mbase = (mt0 + i) * 16 + mhi;
      int nbase = (nt0 + j) * 16 + nlo;
      if (nbase < J1) {
        float* yp = Y + ((size_t)(b * Cc + mbase)) * J1 + nbase;
#pragma unroll
        for (int r = 0; r < 8; ++r) yp[(size_t)r * J1] = acc[r];
      }
    }
  }
}

// ---------------------------------------------------------------------------
// Per-row inverse norms of q and k. One wave per (b,c) row.
// ---------------------------------------------------------------------------
__global__ void norms_kernel(const float* __restrict__ Y,
                             float* __restrict__ invQ,
                             float* __restrict__ invK) {
  const int lane = threadIdx.x & 31;
  const int row  = blockIdx.x * 8 + (threadIdx.x >> 5);   // Bx*Cc rows total
  const float* y = Y + (size_t)row * J1;
  float sq = 0.0f, sk = 0.0f;
  for (int n = lane; n < Nn; n += 32) {
    float q = y[n];       sq += q * q;
    float k = y[Nn + n];  sk += k * k;
  }
#pragma unroll
  for (int off = 16; off > 0; off >>= 1) {
    sq += __shfl_xor(sq, off, 32);
    sk += __shfl_xor(sk, off, 32);
  }
  if (lane == 0) {
    invQ[row] = rsqrtf(sq);
    invK[row] = rsqrtf(sk);
  }
}

// ---------------------------------------------------------------------------
// kernel_trick[b][n] = sum_c  k_hat[b][c][n] * v[b][c][n]
// One block per batch, thread n; fully deterministic (no atomics).
// ---------------------------------------------------------------------------
__global__ void ktrick_kernel(const float* __restrict__ Y,
                              const float* __restrict__ invK,
                              float* __restrict__ KTb) {
  const int b = blockIdx.x;
  const int n = threadIdx.x;
  if (n >= Nn) return;
  const float* y  = Y + (size_t)b * Cc * J1;
  const float* ik = invK + (size_t)b * Cc;
  float acc = 0.0f;
#pragma unroll 4
  for (int c = 0; c < Cc; ++c) {
    const float* row = y + (size_t)c * J1;
    acc += row[Nn + n] * ik[c] * row[2 * Nn + n];
  }
  KTb[(size_t)b * Nn + n] = acc;
}

// ---------------------------------------------------------------------------
// Exact GeLU epilogue + repack to bf16 A-fragments for GEMM2 (aliases Apack).
// ---------------------------------------------------------------------------
__global__ void gelu_pack_kernel(const float* __restrict__ Y,
                                 const float* __restrict__ invQ,
                                 const float* __restrict__ KTb,
                                 const float* __restrict__ scale_p,
                                 v16bf* __restrict__ Gpack) {
  size_t t = (size_t)blockIdx.x * 256 + threadIdx.x;   // Bx*MT*KT*32 threads
  int lane = (int)(t & 31);
  size_t rest = t >> 5;
  int kt = (int)(rest % KT); rest /= KT;
  int mt = (int)(rest % MT);
  int b  = (int)(rest / MT);
  const float scale = scale_p[0];
  int m     = mt * 16 + (lane & 15);
  int kbase = kt * 32 + ((lane >> 4) * 8);
  const float* qrow = Y + ((size_t)(b * Cc + m)) * J1;
  const float  iq   = invQ[(size_t)b * Cc + m];
  const float* ktb  = KTb + (size_t)b * Nn;
  v16bf v;
#pragma unroll
  for (int e = 0; e < 16; ++e) {
    int k = kbase + (e < 8 ? e : e + 8);
    float val = 0.0f;
    if (k < Nn) {
      float xg = qrow[k] * iq * ktb[k] * scale;
      val = 0.5f * xg * (1.0f + erff(xg * 0.70710678118654752f));
    }
    v[e] = (__bf16)val;
  }
  Gpack[t] = v;
}

// ---------------------------------------------------------------------------
// GEMM2: out_t = gelu(...) @ Wproj^T + bproj, stored transposed as (b, n, c).
// ---------------------------------------------------------------------------
__global__ void gemm2_kernel(const v16bf* __restrict__ Gpack,
                             const v16bf* __restrict__ Bpack,
                             const float* __restrict__ bproj,
                             float* __restrict__ Out) {
  const int lane = threadIdx.x & 31;
  const int wave = threadIdx.x >> 5;
  const int mw   = wave & 3;
  const int nw   = wave >> 2;
  const int b    = blockIdx.z;
  const int mt0  = blockIdx.y * 8 + mw * 2;
  const int nt0  = blockIdx.x * 4 + nw * 2;

  v8f acc00 = {}; v8f acc01 = {}; v8f acc10 = {}; v8f acc11 = {};

  const size_t abase0 = ((size_t)(b * MT + mt0)     * KT) * 32 + lane;
  const size_t abase1 = ((size_t)(b * MT + mt0 + 1) * KT) * 32 + lane;
  const size_t bbase0 = ((size_t)(nt0)     * KT) * 32 + lane;
  const size_t bbase1 = ((size_t)(nt0 + 1) * KT) * 32 + lane;

#pragma unroll
  for (int kt = 0; kt < KT; ++kt) {
    v16bf a0 = Gpack[abase0 + (size_t)kt * 32];
    v16bf a1 = Gpack[abase1 + (size_t)kt * 32];
    v16bf b0 = Bpack[bbase0 + (size_t)kt * 32];
    v16bf b1 = Bpack[bbase1 + (size_t)kt * 32];
    acc00 = wmma_bf16(a0, b0, acc00);
    acc01 = wmma_bf16(a0, b1, acc01);
    acc10 = wmma_bf16(a1, b0, acc10);
    acc11 = wmma_bf16(a1, b1, acc11);
  }

  const int nlo = lane & 15;
  const int mhi = (lane >> 4) * 8;
#pragma unroll
  for (int i = 0; i < 2; ++i) {
#pragma unroll
    for (int j = 0; j < 2; ++j) {
      v8f acc = (i == 0) ? (j == 0 ? acc00 : acc01) : (j == 0 ? acc10 : acc11);
      int mbase = (mt0 + i) * 16 + mhi;            // c coordinate
      int nbase = (nt0 + j) * 16 + nlo;            // n coordinate
      if (nbase < Nn) {
        float bias = bproj[nbase];
        // out[b][n][c]: c contiguous -> 8 consecutive floats per lane
        float* op = Out + ((size_t)b * Nn + nbase) * Cc + mbase;
#pragma unroll
        for (int r = 0; r < 8; ++r) op[r] = acc[r] + bias;
      }
    }
  }
}

// ---------------------------------------------------------------------------
extern "C" void kernel_launch(void* const* d_in, const int* in_sizes, int n_in,
                              void* d_out, int out_size, void* d_ws, size_t ws_size,
                              hipStream_t stream) {
  const float* x     = (const float*)d_in[0];
  const float* Wq    = (const float*)d_in[1];
  const float* Wkv   = (const float*)d_in[2];
  const float* Wproj = (const float*)d_in[3];
  const float* bproj = (const float*)d_in[4];
  const float* scale = (const float*)d_in[5];
  float* out = (float*)d_out;

  char* ws = (char*)d_ws;
  size_t off = 0;
  const size_t APACK_BYTES = (size_t)Bx * MT * KT * 32 * sizeof(v16bf); // 22 MB
  const size_t BP1_BYTES   = (size_t)NT1 * KT * 32 * sizeof(v16bf);
  const size_t BPP_BYTES   = (size_t)NT2 * KT * 32 * sizeof(v16bf);
  const size_t Y_BYTES     = (size_t)Bx * Cc * J1 * sizeof(float);      // 116 MB
  const size_t IV_BYTES    = (size_t)Bx * Cc * sizeof(float);
  const size_t KT_BYTES    = (size_t)Bx * Nn * sizeof(float);

  v16bf* Apack  = (v16bf*)(ws + off); off += APACK_BYTES;   // reused as Gpack
  v16bf* Bpack1 = (v16bf*)(ws + off); off += BP1_BYTES;
  v16bf* BpackP = (v16bf*)(ws + off); off += BPP_BYTES;
  float* Y      = (float*)(ws + off); off += Y_BYTES;
  float* invQ   = (float*)(ws + off); off += IV_BYTES;
  float* invK   = (float*)(ws + off); off += IV_BYTES;
  float* KTb    = (float*)(ws + off); off += KT_BYTES;

  const int packx_blocks = (Bx * MT * KT * 32) / 256;     // 2688
  pack_x_kernel<<<packx_blocks, 256, 0, stream>>>(x, Apack);
  pack_w1_kernel<<<(NT1 * KT * 32 + 255) / 256, 256, 0, stream>>>(Wq, Wkv, Bpack1);
  pack_wp_kernel<<<(NT2 * KT * 32 + 255) / 256, 256, 0, stream>>>(Wproj, BpackP);

  gemm1_kernel<<<dim3(NT1 / 4, MT / 8, Bx), 256, 0, stream>>>(Apack, Bpack1, Y);

  norms_kernel<<<(Bx * Cc) / 8, 256, 0, stream>>>(Y, invQ, invK);
  ktrick_kernel<<<Bx, 224, 0, stream>>>(Y, invK, KTb);

  // Gpack aliases Apack (GEMM1 already consumed it; stream-ordered = safe)
  gelu_pack_kernel<<<packx_blocks, 256, 0, stream>>>(Y, invQ, KTb, scale, Apack);

  gemm2_kernel<<<dim3(NT2 / 4, MT / 8, Bx), 256, 0, stream>>>(Apack, BpackP, bproj, out);
}
